// W8A8Linear_1657857376643
// MI455X (gfx1250) — 8-bit matrix operand order repaired and checked by reading, compiled, not run on hardware
//
#include <hip/hip_runtime.h>
#include <hip/hip_bf16.h>
#include <stdint.h>

typedef __attribute__((ext_vector_type(8))) int v8i_t;

#define QTHREADS 256
#define QPER     16

__global__ __launch_bounds__(QTHREADS)
void w8a8_quant_kernel(const float* __restrict__ x,
                       const float* __restrict__ smooth,
                       int8_t* __restrict__ q,
                       float* __restrict__ xscale,
                       int K)
{
    const int row = blockIdx.x;
    const int tid = threadIdx.x;
    const float* xr  = x + (size_t)row * K;
    const int   base = tid * QPER;

    float y[QPER];
    float amax = 0.0f;
#pragma unroll
    for (int j = 0; j < QPER; j += 4) {
        float4 xv = *(const float4*)(xr + base + j);
        float4 sv = *(const float4*)(smooth + base + j);
        y[j + 0] = xv.x / sv.x;
        y[j + 1] = xv.y / sv.y;
        y[j + 2] = xv.z / sv.z;
        y[j + 3] = xv.w / sv.w;
        amax = fmaxf(amax, fabsf(y[j + 0]));
        amax = fmaxf(amax, fabsf(y[j + 1]));
        amax = fmaxf(amax, fabsf(y[j + 2]));
        amax = fmaxf(amax, fabsf(y[j + 3]));
    }

#pragma unroll
    for (int off = 16; off > 0; off >>= 1)
        amax = fmaxf(amax, __shfl_xor(amax, off, 32));

    __shared__ float smax[QTHREADS / 32];
    if ((tid & 31) == 0) smax[tid >> 5] = amax;
    __syncthreads();
    float m = smax[0];
#pragma unroll
    for (int i = 1; i < QTHREADS / 32; ++i) m = fmaxf(m, smax[i]);

    const float mm    = fmaxf(m, 1e-7f);
    const float scale = mm * (1.0f / 127.0f);
    const float inv   = 127.0f / mm;
    if (tid == 0) xscale[row] = scale;

    uint32_t wrd[QPER / 4];
#pragma unroll
    for (int g = 0; g < QPER / 4; ++g) {
        uint32_t wv = 0u;
#pragma unroll
        for (int j = 0; j < 4; ++j) {
            int iv = (int)rintf(y[g * 4 + j] * inv);
            iv = iv > 127 ? 127 : (iv < -127 ? -127 : iv);
            wv |= ((uint32_t)(uint8_t)(int8_t)iv) << (8 * j);
        }
        wrd[g] = wv;
    }
    *(int4*)(q + (size_t)row * K + base) =
        make_int4((int)wrd[0], (int)wrd[1], (int)wrd[2], (int)wrd[3]);
}

__global__ __launch_bounds__(256)
void w8a8_wmma_gemm_kernel(const int8_t* __restrict__ qa,
                           const int8_t* __restrict__ wb,
                           const float*  __restrict__ xscale,
                           const float*  __restrict__ wscale,
                           const float*  __restrict__ bias,
                           __hip_bfloat16* __restrict__ out,
                           int M, int N, int K)
{
    const int tid    = threadIdx.x;
    const int wave   = tid >> 5;
    const int lane   = tid & 31;
    const int half   = lane >> 4;
    const int l16    = lane & 15;
    const int wave_m = wave >> 2;
    const int wave_n = wave & 3;

    const int m0 = blockIdx.y * 128 + wave_m * 64;
    const int n0 = blockIdx.x * 128 + wave_n * 32;

    const int8_t* arow[4];
#pragma unroll
    for (int t = 0; t < 4; ++t)
        arow[t] = qa + (size_t)(m0 + t * 16 + l16) * K + half * 16;

    const int8_t* brow[2];
#pragma unroll
    for (int u = 0; u < 2; ++u)
        brow[u] = wb + (size_t)(n0 + u * 16 + l16) * K + half * 16;

    v8i_t acc[4][2];
#pragma unroll
    for (int t = 0; t < 4; ++t)
#pragma unroll
        for (int u = 0; u < 2; ++u)
#pragma unroll
            for (int r = 0; r < 8; ++r) acc[t][u][r] = 0;

    for (int k0 = 0; k0 < K; k0 += 64) {
        v8i_t bfrag[2];
#pragma unroll
        for (int u = 0; u < 2; ++u) {
            int4 lo = *(const int4*)(brow[u] + k0);
            int4 hi = *(const int4*)(brow[u] + k0 + 32);
            bfrag[u][0] = lo.x; bfrag[u][1] = lo.y;
            bfrag[u][2] = lo.z; bfrag[u][3] = lo.w;
            bfrag[u][4] = hi.x; bfrag[u][5] = hi.y;
            bfrag[u][6] = hi.z; bfrag[u][7] = hi.w;
        }
        v8i_t afrag[4];
#pragma unroll
        for (int t = 0; t < 4; ++t) {
#pragma unroll
            for (int blk = 0; blk < 4; ++blk) {
                uint2 p = *(const uint2*)(arow[t] + k0 + (blk >> 1) * 32 + (blk & 1) * 8);
                afrag[t][2 * blk]     = (int)p.x;
                afrag[t][2 * blk + 1] = (int)p.y;
            }
        }
        if (k0 + 256 < K) {
            __builtin_prefetch(arow[0] + k0 + 256, 0, 1);
            __builtin_prefetch(brow[0] + k0 + 256, 0, 1);
        }
#pragma unroll
        for (int t = 0; t < 4; ++t)
#pragma unroll
            for (int u = 0; u < 2; ++u)
                acc[t][u] = __builtin_amdgcn_wmma_i32_16x16x64_iu8(
                    true, afrag[t], true, bfrag[u], acc[t][u], false, false);
    }

#pragma unroll
    for (int t = 0; t < 4; ++t) {
        float xs[8];
#pragma unroll
        for (int r = 0; r < 8; ++r)
            xs[r] = xscale[m0 + t * 16 + r + 8 * half];
#pragma unroll
        for (int u = 0; u < 2; ++u) {
            const int   n   = n0 + u * 16 + l16;
            const float wsn = wscale[n];
            const float bn  = bias[n];
#pragma unroll
            for (int r = 0; r < 8; ++r) {
                const int m = m0 + t * 16 + r + 8 * half;
                const float v = (float)acc[t][u][r] * xs[r] * wsn + bn;
                out[(size_t)m * N + n] = __float2bfloat16(v);
            }
        }
    }
}

extern "C" void kernel_launch(void* const* d_in, const int* in_sizes, int n_in,
                              void* d_out, int out_size, void* d_ws, size_t ws_size,
                              hipStream_t stream)
{
    const float*  x      = (const float*) d_in[0];
    const int8_t* weight = (const int8_t*)d_in[1];
    const float*  wscale = (const float*) d_in[2];
    const float*  smooth = (const float*) d_in[3];
    const float*  bias   = (const float*) d_in[4];

    const int K = in_sizes[3];
    const int N = in_sizes[4];
    const int M = in_sizes[0] / K;

    int8_t* q      = (int8_t*)d_ws;
    float*  xscale = (float*)((char*)d_ws + (size_t)M * (size_t)K);

    __hip_bfloat16* out = (__hip_bfloat16*)d_out;

    w8a8_quant_kernel<<<M, QTHREADS, 0, stream>>>(x, smooth, q, xscale, K);

    dim3 grid(N / 128, M / 128);
    w8a8_wmma_gemm_kernel<<<grid, 256, 0, stream>>>(q, weight, xscale, wscale,
                                                    bias, out, M, N, K);
}
